// PGN_19980187861568
// MI455X (gfx1250) — matrix operand order repaired and checked by reading, not compiled, not run on hardware
//
#include <hip/hip_runtime.h>
#include <math.h>

#define BB   64
#define LQ_  512
#define LC_  2048
#define DD   256
#define DV_  50257
#define MOOV_ 512
#define DVE  (DV_ + MOOV_)
#define NVT  ((DV_ + 15) / 16)   // 3142 vocab tiles

typedef __attribute__((ext_vector_type(16))) _Float16 v16h;
typedef __attribute__((ext_vector_type(8)))  float    v8f;

// gfx1250 has V_TANH_F32; use it if clang exposes the builtin.
#if defined(__has_builtin)
#if __has_builtin(__builtin_amdgcn_tanhf)
#define FAST_TANH(x) __builtin_amdgcn_tanhf(x)
#endif
#endif
#ifndef FAST_TANH
#define FAST_TANH(x) tanhf(x)
#endif

// A-fragment K index for 16-bit 16x32 A matrix (contiguous order, matches B):
// lanes 0-15 (hv=0): K=0..15
// lanes 16-31 (hv=1): K=16..31
__device__ __forceinline__ int a_kmap(int e, int hv) {
  return (hv << 4) + e;
}

__device__ __forceinline__ float sigm(float x) { return 1.0f / (1.0f + __expf(-x)); }

// ---------- one-time f32 -> f16 conversion of W_att1 and W_ff1 ----------
__global__ void k_cvtw(const float* __restrict__ W_att1, const float* __restrict__ W_ff1,
                       _Float16* __restrict__ w1h, _Float16* __restrict__ wff1h) {
  int idx = blockIdx.x * 256 + threadIdx.x;      // 2 * 65536 elements
  if (idx < DD * DD) w1h[idx] = (_Float16)W_att1[idx];
  else wff1h[idx - DD * DD] = (_Float16)W_ff1[idx - DD * DD];
}

// ---------- s1[b,e] = ai[b,:] . W_att2[e,:] + b_att2[e] ----------
__global__ void k_s1(const float* __restrict__ ai, const float* __restrict__ W2,
                     const float* __restrict__ b2, float* __restrict__ s1) {
  int b = blockIdx.x, e = threadIdx.x;
  __shared__ float al[DD];
  al[e] = ai[b * DD + e];
  __syncthreads();
  const float* wr = W2 + (size_t)e * DD;
  float s = b2[e];
  for (int d = 0; d < DD; ++d) s += al[d] * wr[d];
  s1[b * DD + e] = s;
}

// ---------- a[b,l] = w_attn . tanh(q W1^T + s1) + b_attn  (WMMA, fused) ----------
__global__ void __launch_bounds__(256) k_qscore(
    const float* __restrict__ q, const _Float16* __restrict__ W1h,
    const float* __restrict__ s1, const float* __restrict__ w_attn,
    const float* __restrict__ b_attn, float* __restrict__ a_out) {
  __shared__ _Float16 Wl[DD * DD];   // 128 KB f16 W_att1
  __shared__ float s1l[DD], wal[DD];
  int tid = threadIdx.x;
  int rows0 = blockIdx.x * 128;      // 128 q-rows per block; never crosses b (512/128)
  int b = rows0 / LQ_;
  {
    uint4* dst = (uint4*)Wl;
    const uint4* src = (const uint4*)W1h;
    for (int i = tid; i < (DD * DD) / 8; i += 256) dst[i] = src[i];
  }
  s1l[tid] = s1[b * DD + tid];
  wal[tid] = w_attn[tid];
  __syncthreads();
  int wave = tid >> 5, lane = tid & 31, lr = lane & 15, hv = lane >> 4;
  int row = rows0 + wave * 16 + lr;
  const float* qrow = q + (size_t)row * DD;
  v16h afr[8];
#pragma unroll
  for (int ks = 0; ks < 8; ++ks)
#pragma unroll
    for (int e = 0; e < 16; ++e)
      afr[ks][e] = (_Float16)qrow[ks * 32 + a_kmap(e, hv)];
  float partial[8] = {0.f,0.f,0.f,0.f,0.f,0.f,0.f,0.f};
  for (int et = 0; et < 16; ++et) {
    int ecol = et * 16 + lr;
    v8f acc = {};
#pragma unroll
    for (int ks = 0; ks < 8; ++ks) {
      v16h bfr = *(const v16h*)(&Wl[ecol * DD + ks * 32 + hv * 16]);
      acc = __builtin_amdgcn_wmma_f32_16x16x32_f16(false, afr[ks], false, bfr,
                                                   (short)0, acc, false, false);
    }
    float we = wal[ecol], sv = s1l[ecol];
#pragma unroll
    for (int r = 0; r < 8; ++r) partial[r] += we * FAST_TANH(acc[r] + sv);
  }
#pragma unroll
  for (int r = 0; r < 8; ++r) {
    partial[r] += __shfl_xor(partial[r], 1, 32);
    partial[r] += __shfl_xor(partial[r], 2, 32);
    partial[r] += __shfl_xor(partial[r], 4, 32);
    partial[r] += __shfl_xor(partial[r], 8, 32);
  }
  if (lr == 0) {
    float ba = b_attn[0];
    int rbase = rows0 + wave * 16 + hv * 8;
#pragma unroll
    for (int r = 0; r < 8; ++r) a_out[rbase + r] = partial[r] + ba;
  }
}

// ---------- masked softmax over LQ (normalize in place) + zero pooled ----------
__global__ void k_qsm(float* a_q, const unsigned char* __restrict__ qmask,
                      float* __restrict__ pooled) {
  int b = blockIdx.x, tid = threadIdx.x;
  __shared__ float av[LQ_];
  __shared__ float red[256];
  float m = -1e30f;
  for (int l = tid; l < LQ_; l += 256) {
    float v = qmask[b * LQ_ + l] ? -1e30f : a_q[b * LQ_ + l];
    av[l] = v; m = fmaxf(m, v);
  }
  red[tid] = m; __syncthreads();
  for (int s = 128; s > 0; s >>= 1) { if (tid < s) red[tid] = fmaxf(red[tid], red[tid + s]); __syncthreads(); }
  m = red[0]; __syncthreads();
  float ss = 0.f;
  for (int l = tid; l < LQ_; l += 256) { float e = __expf(av[l] - m); av[l] = e; ss += e; }
  red[tid] = ss; __syncthreads();
  for (int s = 128; s > 0; s >>= 1) { if (tid < s) red[tid] += red[tid + s]; __syncthreads(); }
  float inv = 1.0f / red[0]; __syncthreads();
  for (int l = tid; l < LQ_; l += 256) a_q[b * LQ_ + l] = av[l] * inv;
  pooled[b * DD + tid] = 0.f;
}

// ---------- pooled = a . q  (chunked over l for parallelism) ----------
__global__ void k_pool(const float* __restrict__ a_q, const float* __restrict__ q,
                       float* __restrict__ pooled) {
  int cid = blockIdx.x;                 // BB * (LQ/256) = 128 blocks
  int b = cid / (LQ_ / 256), l0 = (cid % (LQ_ / 256)) * 256;
  int tid = threadIdx.x;                // d
  __shared__ float al[256];
  al[tid] = a_q[b * LQ_ + l0 + tid];
  __syncthreads();
  float acc = 0.f;
  const float* qb = q + ((size_t)b * LQ_ + l0) * DD + tid;
  for (int l = 0; l < 256; ++l) acc += al[l] * qb[(size_t)l * DD];
  atomicAdd(&pooled[b * DD + tid], acc);
}

// ---------- at_raw[b,l] = w_ff2 . tanh(x W_ff1^T + b_ff1) + b_ff2  (WMMA, fused) ----------
__global__ void __launch_bounds__(256) k_cscore(
    const float* __restrict__ Y, const _Float16* __restrict__ W1h,
    const float* __restrict__ h, const float* __restrict__ pooled,
    const float* __restrict__ cov, const float* __restrict__ w_cov,
    const float* __restrict__ b_ff1, const float* __restrict__ w_ff2,
    const float* __restrict__ b_ff2, float* __restrict__ at_raw) {
  __shared__ _Float16 Wl[DD * DD];   // 128 KB f16 W_ff1
  __shared__ float basel[DD], wcl[DD], b1l[DD], w2l[DD];
  int tid = threadIdx.x;
  int rows0 = blockIdx.x * 128;      // 128 of the B*LC rows; 2048%128==0 -> one b per block
  int b = rows0 / LC_;
  {
    uint4* dst = (uint4*)Wl;
    const uint4* src = (const uint4*)W1h;
    for (int i = tid; i < (DD * DD) / 8; i += 256) dst[i] = src[i];
  }
  basel[tid] = h[b * DD + tid] + pooled[b * DD + tid];
  wcl[tid] = w_cov[tid];
  b1l[tid] = b_ff1[tid];
  w2l[tid] = w_ff2[tid];
  __syncthreads();
  int wave = tid >> 5, lane = tid & 31, lr = lane & 15, hv = lane >> 4;
  int row = rows0 + wave * 16 + lr;             // flat index into [B*LC)
  const float* Yrow = Y + (size_t)row * DD;
  float covv = cov[row];
  v16h afr[8];
#pragma unroll
  for (int ks = 0; ks < 8; ++ks)
#pragma unroll
    for (int e = 0; e < 16; ++e) {
      int dk = ks * 32 + a_kmap(e, hv);
      afr[ks][e] = (_Float16)(Yrow[dk] + basel[dk] + covv * wcl[dk]);
    }
  float partial[8] = {0.f,0.f,0.f,0.f,0.f,0.f,0.f,0.f};
  for (int et = 0; et < 16; ++et) {
    int ecol = et * 16 + lr;
    v8f acc = {};
#pragma unroll
    for (int ks = 0; ks < 8; ++ks) {
      v16h bfr = *(const v16h*)(&Wl[ecol * DD + ks * 32 + hv * 16]);
      acc = __builtin_amdgcn_wmma_f32_16x16x32_f16(false, afr[ks], false, bfr,
                                                   (short)0, acc, false, false);
    }
    float we = w2l[ecol], bv = b1l[ecol];
#pragma unroll
    for (int r = 0; r < 8; ++r) partial[r] += we * FAST_TANH(acc[r] + bv);
  }
#pragma unroll
  for (int r = 0; r < 8; ++r) {
    partial[r] += __shfl_xor(partial[r], 1, 32);
    partial[r] += __shfl_xor(partial[r], 2, 32);
    partial[r] += __shfl_xor(partial[r], 4, 32);
    partial[r] += __shfl_xor(partial[r], 8, 32);
  }
  if (lr == 0) {
    float bf = b_ff2[0];
    int rbase = rows0 + wave * 16 + hv * 8;
#pragma unroll
    for (int r = 0; r < 8; ++r) at_raw[rbase + r] = partial[r] + bf;
  }
}

// ---------- masked softmax over LC (in place) + cov_new + zero yv ----------
__global__ void k_csm(float* at, const unsigned char* __restrict__ cmask,
                      const float* __restrict__ cov, float* __restrict__ cov_new,
                      float* __restrict__ yv) {
  int b = blockIdx.x, tid = threadIdx.x;
  __shared__ float ex[LC_];
  __shared__ float red[256];
  float m = -1e30f;
  for (int l = tid; l < LC_; l += 256) {
    float v = cmask[b * LC_ + l] ? -1e30f : at[b * LC_ + l];
    ex[l] = v; m = fmaxf(m, v);
  }
  red[tid] = m; __syncthreads();
  for (int s = 128; s > 0; s >>= 1) { if (tid < s) red[tid] = fmaxf(red[tid], red[tid + s]); __syncthreads(); }
  m = red[0]; __syncthreads();
  float ss = 0.f;
  for (int l = tid; l < LC_; l += 256) { float e = __expf(ex[l] - m); ex[l] = e; ss += e; }
  red[tid] = ss; __syncthreads();
  for (int s = 128; s > 0; s >>= 1) { if (tid < s) red[tid] += red[tid + s]; __syncthreads(); }
  float inv = 1.0f / red[0]; __syncthreads();
  for (int l = tid; l < LC_; l += 256) {
    float a = ex[l] * inv;
    at[b * LC_ + l] = a;
    cov_new[b * LC_ + l] = cov[b * LC_ + l] + a;
  }
  yv[b * DD + tid] = 0.f;
}

// ---------- y = at . Y  (chunked over l for parallelism) ----------
__global__ void k_ydot(const float* __restrict__ at, const float* __restrict__ Y,
                       float* __restrict__ yv) {
  int cid = blockIdx.x;                 // BB * (LC/256) = 512 blocks
  int b = cid / (LC_ / 256), l0 = (cid % (LC_ / 256)) * 256;
  int tid = threadIdx.x;                // d
  __shared__ float al[256];
  al[tid] = at[b * LC_ + l0 + tid];
  __syncthreads();
  float acc = 0.f;
  const float* Yb = Y + ((size_t)b * LC_ + l0) * DD + tid;
  for (int l = 0; l < 256; ++l) acc += al[l] * Yb[(size_t)l * DD];
  atomicAdd(&yv[b * DD + tid], acc);
}

// ---------- LSTM cell + pt ----------
__global__ void k_lstm(const float* __restrict__ yv, const float* __restrict__ ai,
                       const float* __restrict__ h, const float* __restrict__ cprev,
                       const float* __restrict__ W_ih, const float* __restrict__ W_hh,
                       const float* __restrict__ b_ih, const float* __restrict__ b_hh,
                       const float* __restrict__ w_p1, const float* __restrict__ w_p2,
                       const float* __restrict__ w_p3, float* __restrict__ cnew,
                       float* __restrict__ hnew, float* __restrict__ pt) {
  int b = blockIdx.x, tid = threadIdx.x;
  __shared__ float yl[DD], al[DD], hl[DD], red[256];
  yl[tid] = yv[b * DD + tid]; al[tid] = ai[b * DD + tid]; hl[tid] = h[b * DD + tid];
  __syncthreads();
  float g[4];
#pragma unroll
  for (int gi = 0; gi < 4; ++gi) {
    int j = gi * DD + tid;
    const float* wi = W_ih + (size_t)j * (2 * DD);
    const float* wh = W_hh + (size_t)j * DD;
    float s = b_ih[j] + b_hh[j];
    for (int d = 0; d < DD; ++d) s += yl[d] * wi[d] + al[d] * wi[DD + d] + hl[d] * wh[d];
    g[gi] = s;
  }
  float cn = sigm(g[1]) * cprev[b * DD + tid] + sigm(g[0]) * FAST_TANH(g[2]);
  float hn = sigm(g[3]) * FAST_TANH(cn);
  cnew[b * DD + tid] = cn; hnew[b * DD + tid] = hn;
  red[tid] = cn * w_p1[tid] + hn * w_p2[tid] + yl[tid] * w_p3[tid];
  __syncthreads();
  for (int s = 128; s > 0; s >>= 1) { if (tid < s) red[tid] += red[tid + s]; __syncthreads(); }
  if (tid == 0) pt[b] = sigm(red[0]);
}

// ---------- logits = h_new W_pgn^T + b_pgn  (WMMA; W_pgn streamed once) ----------
__global__ void __launch_bounds__(256) k_pgn(
    const float* __restrict__ hnew, const float* __restrict__ W_pgn,
    const float* __restrict__ b_pgn, float* __restrict__ logits) {
  __shared__ _Float16 h16[BB * DD];   // 32 KB
  int tid = threadIdx.x;
  for (int i = tid; i < BB * DD; i += 256) h16[i] = (_Float16)hnew[i];
  __syncthreads();
  int wave = tid >> 5, lane = tid & 31, lr = lane & 15, hv = lane >> 4;
  int vt = blockIdx.x * 8 + wave;
  if (vt >= NVT) return;              // wave-uniform: EXEC stays all-1 for WMMA
  int vcol = vt * 16 + lr;
  int vc = vcol < DV_ ? vcol : DV_ - 1;
  v8f acc[4] = {};
  const float* wrow = W_pgn + (size_t)vc * DD;
#pragma unroll
  for (int ks = 0; ks < 8; ++ks) {
    v16h bfr;
    const float* wp = wrow + ks * 32 + hv * 16;
#pragma unroll
    for (int e = 0; e < 16; ++e) bfr[e] = (_Float16)wp[e];
#pragma unroll
    for (int mt = 0; mt < 4; ++mt) {
      v16h afr;
      const _Float16* hb = &h16[(mt * 16 + lr) * DD + ks * 32];
#pragma unroll
      for (int e = 0; e < 16; ++e) afr[e] = hb[a_kmap(e, hv)];
      acc[mt] = __builtin_amdgcn_wmma_f32_16x16x32_f16(false, afr, false, bfr,
                                                       (short)0, acc[mt], false, false);
    }
  }
  if (vcol < DV_) {
    float bp = b_pgn[vcol];
#pragma unroll
    for (int mt = 0; mt < 4; ++mt)
#pragma unroll
      for (int r = 0; r < 8; ++r) {
        int bi = mt * 16 + hv * 8 + r;
        logits[(size_t)bi * DV_ + vcol] = acc[mt][r] + bp;
      }
  }
}

// ---------- per-row max & sum(exp) over DV ----------
__global__ void k_rowstat(const float* __restrict__ logits,
                          float* __restrict__ rmax, float* __restrict__ rsum) {
  int b = blockIdx.x, tid = threadIdx.x;
  __shared__ float red[256];
  const float* lb = logits + (size_t)b * DV_;
  float m = -1e30f;
  for (int v = tid; v < DV_; v += 256) m = fmaxf(m, lb[v]);
  red[tid] = m; __syncthreads();
  for (int s = 128; s > 0; s >>= 1) { if (tid < s) red[tid] = fmaxf(red[tid], red[tid + s]); __syncthreads(); }
  m = red[0]; __syncthreads();
  float ss = 0.f;
  for (int v = tid; v < DV_; v += 256) ss += __expf(lb[v] - m);
  red[tid] = ss; __syncthreads();
  for (int s = 128; s > 0; s >>= 1) { if (tid < s) red[tid] += red[tid + s]; __syncthreads(); }
  if (tid == 0) { rmax[b] = m; rsum[b] = red[0]; }
}

// ---------- wt base: (1-pt)*softmax for v<DV, zeros for OOV slots ----------
__global__ void k_wt(const float* __restrict__ logits, const float* __restrict__ rmax,
                     const float* __restrict__ rsum, const float* __restrict__ pt,
                     float* __restrict__ wt) {
  long long idx = (long long)blockIdx.x * 256 + threadIdx.x;
  if (idx >= (long long)BB * DVE) return;
  int b = (int)(idx / DVE), v = (int)(idx % DVE);
  float out = 0.f;
  if (v < DV_) out = (1.0f - pt[b]) * __expf(logits[(size_t)b * DV_ + v] - rmax[b]) / rsum[b];
  wt[idx] = out;
}

// ---------- copy mechanism: at_c + scatter-add ----------
__global__ void k_scatter(const float* __restrict__ at, const float* __restrict__ pt,
                          const int* __restrict__ ids, float* __restrict__ wt,
                          float* __restrict__ at_c) {
  int idx = blockIdx.x * 256 + threadIdx.x;   // < B*LC
  int b = idx / LC_;
  float v = pt[b] * at[idx];
  at_c[idx] = v;
  atomicAdd(&wt[(size_t)b * DVE + ids[idx]], v);
}

extern "C" void kernel_launch(void* const* d_in, const int* in_sizes, int n_in,
                              void* d_out, int out_size, void* d_ws, size_t ws_size,
                              hipStream_t stream) {
  (void)in_sizes; (void)n_in; (void)out_size; (void)ws_size;
  const float* ai     = (const float*)d_in[0];
  const float* h      = (const float*)d_in[1];
  const float* c      = (const float*)d_in[2];
  const float* cov    = (const float*)d_in[3];
  const float* Y      = (const float*)d_in[4];
  const float* q      = (const float*)d_in[5];
  const unsigned char* q_masks = (const unsigned char*)d_in[6];  // jnp bool = 1 byte
  const int*   c_ids  = (const int*)d_in[7];
  const unsigned char* c_masks = (const unsigned char*)d_in[8];
  // d_in[9] = max_oov_len (compile-time constant MOOV_ used)
  const float* W_att1 = (const float*)d_in[10];
  const float* W_att2 = (const float*)d_in[11];
  const float* b_att2 = (const float*)d_in[12];
  const float* w_attn = (const float*)d_in[13];
  const float* b_attn = (const float*)d_in[14];
  const float* W_ff1  = (const float*)d_in[15];
  const float* b_ff1  = (const float*)d_in[16];
  const float* w_ff2  = (const float*)d_in[17];
  const float* b_ff2  = (const float*)d_in[18];
  const float* w_cov  = (const float*)d_in[19];
  const float* W_ih   = (const float*)d_in[20];
  const float* W_hh   = (const float*)d_in[21];
  const float* b_ih   = (const float*)d_in[22];
  const float* b_hh   = (const float*)d_in[23];
  const float* W_pgn  = (const float*)d_in[24];
  const float* b_pgn  = (const float*)d_in[25];
  const float* w_p1   = (const float*)d_in[26];
  const float* w_p2   = (const float*)d_in[27];
  const float* w_p3   = (const float*)d_in[28];

  float* wt      = (float*)d_out;
  float* cov_new = wt + (size_t)BB * DVE;
  float* at_c    = cov_new + (size_t)BB * LC_;

  float* ws = (float*)d_ws;
  float* s1     = ws; ws += BB * DD;
  float* a_q    = ws; ws += BB * LQ_;
  float* pooled = ws; ws += BB * DD;
  float* at     = ws; ws += BB * LC_;
  float* yv     = ws; ws += BB * DD;
  float* cnew   = ws; ws += BB * DD;
  float* hnew   = ws; ws += BB * DD;
  float* pt     = ws; ws += BB;
  float* rmax   = ws; ws += BB;
  float* rsum   = ws; ws += BB;
  float* logits = ws; ws += (size_t)BB * DV_;
  _Float16* w1h   = (_Float16*)ws;            // 65536 f16 = 32768 floats
  _Float16* wff1h = w1h + DD * DD;            // 65536 f16

  k_cvtw<<<(2 * DD * DD) / 256, 256, 0, stream>>>(W_att1, W_ff1, w1h, wff1h);
  k_s1<<<BB, 256, 0, stream>>>(ai, W_att2, b_att2, s1);
  k_qscore<<<(BB * LQ_) / 128, 256, 0, stream>>>(q, w1h, s1, w_attn, b_attn, a_q);
  k_qsm<<<BB, 256, 0, stream>>>(a_q, q_masks, pooled);
  k_pool<<<BB * (LQ_ / 256), 256, 0, stream>>>(a_q, q, pooled);
  k_cscore<<<(BB * LC_) / 128, 256, 0, stream>>>(Y, wff1h, h, pooled, cov, w_cov,
                                                 b_ff1, w_ff2, b_ff2, at);
  k_csm<<<BB, 256, 0, stream>>>(at, c_masks, cov, cov_new, yv);
  k_ydot<<<BB * (LC_ / 256), 256, 0, stream>>>(at, Y, yv);
  k_lstm<<<BB, 256, 0, stream>>>(yv, ai, h, c, W_ih, W_hh, b_ih, b_hh,
                                 w_p1, w_p2, w_p3, cnew, hnew, pt);
  k_pgn<<<(NVT + 7) / 8, 256, 0, stream>>>(hnew, W_pgn, b_pgn, logits);
  k_rowstat<<<BB, 256, 0, stream>>>(logits, rmax, rsum);
  k_wt<<<(int)(((long long)BB * DVE + 255) / 256), 256, 0, stream>>>(logits, rmax, rsum, pt, wt);
  k_scatter<<<(BB * LC_) / 256, 256, 0, stream>>>(at, pt, c_ids, wt, at_c);
}
